// PokemonAttention_23141283791042
// MI455X (gfx1250) — matrix operand order repaired and checked by reading, not compiled, not run on hardware
//
#include <hip/hip_runtime.h>

typedef __attribute__((ext_vector_type(16))) __bf16          v16bf;
typedef __attribute__((ext_vector_type(8)))  float           v8f;
typedef __attribute__((ext_vector_type(8)))  unsigned short  u16x8;
typedef __attribute__((ext_vector_type(4)))  unsigned short  u16x4;
typedef __attribute__((ext_vector_type(4)))  float           f32x4;

__device__ __forceinline__ unsigned short f2bf(float f) {
    unsigned int u = __float_as_uint(f);
    u += 0x7FFFu + ((u >> 16) & 1u);   // round-to-nearest-even
    return (unsigned short)(u >> 16);
}

union BF16Frag { v16bf v; unsigned short u[16]; u16x8 h[2]; };

// ---------------------------------------------------------------------------
// fp32 -> bf16 bulk convert, 4 elements/thread
// ---------------------------------------------------------------------------
__global__ __launch_bounds__(256) void tobf16_kernel(const float* __restrict__ in,
                                                     unsigned short* __restrict__ out,
                                                     int n4) {
    const int t = blockIdx.x * 256 + threadIdx.x;
    if (t >= n4) return;
    f32x4 f = *(const f32x4*)(in + (size_t)t * 4);
    u16x4 o;
    for (int i = 0; i < 4; ++i) o[i] = f2bf(f[i]);
    *(u16x4*)(out + (size_t)t * 4) = o;
}

// ---------------------------------------------------------------------------
// GEMM: C[M,N] = A[M,K] @ W[N,K]^T, A/W bf16, fp32 accumulate via
// v_wmma_f32_16x16x32_bf16. Block = 256 (8 waves); wave -> 32 rows x 64 cols
// (2 A-frags x 4 B-frags = 8 accumulators, B reused twice). Software-
// pipelined: next k-step's 12 b128 loads issued before current 8 WMMAs.
// ---------------------------------------------------------------------------
__global__ __launch_bounds__(256) void gemm_wmma(const unsigned short* __restrict__ A,
                                                 const unsigned short* __restrict__ W,
                                                 float* __restrict__ C,
                                                 int M, int N, int K) {
    const int lane = threadIdx.x & 31, wave = threadIdx.x >> 5;
    const int l16 = lane & 15, half = lane >> 4;
    const int m0 = blockIdx.y * 256 + wave * 32;
    const int n0 = blockIdx.x * 64;

    v8f acc[2][4];
    for (int mi = 0; mi < 2; ++mi)
        for (int j = 0; j < 4; ++j)
            for (int i = 0; i < 8; ++i) acc[mi][j][i] = 0.0f;

    const int kb1 = half * 16;      // A K-offset per half (K 0-15 / 16-31)
    const int kb2 = half * 16;      // B K-offset per half (K 0-15 / 16-31)

    const unsigned short* pa0 = A + (size_t)(m0 + l16) * K + kb1;
    const unsigned short* pa1 = A + (size_t)(m0 + 16 + l16) * K + kb1;
    const unsigned short* pw[4];
    for (int j = 0; j < 4; ++j)
        pw[j] = W + (size_t)(n0 + j * 16 + l16) * K + kb2;

    BF16Frag a0c, a1c, bc[4];
    a0c.h[0] = *(const u16x8*)pa0;        a0c.h[1] = *(const u16x8*)(pa0 + 8);
    a1c.h[0] = *(const u16x8*)pa1;        a1c.h[1] = *(const u16x8*)(pa1 + 8);
    for (int j = 0; j < 4; ++j) {
        bc[j].h[0] = *(const u16x8*)pw[j];
        bc[j].h[1] = *(const u16x8*)(pw[j] + 8);
    }

    for (int kt = 32; kt < K; kt += 32) {
        BF16Frag a0n, a1n, bn[4];
        a0n.h[0] = *(const u16x8*)(pa0 + kt);  a0n.h[1] = *(const u16x8*)(pa0 + kt + 8);
        a1n.h[0] = *(const u16x8*)(pa1 + kt);  a1n.h[1] = *(const u16x8*)(pa1 + kt + 8);
        for (int j = 0; j < 4; ++j) {
            bn[j].h[0] = *(const u16x8*)(pw[j] + kt);
            bn[j].h[1] = *(const u16x8*)(pw[j] + kt + 8);
        }
        for (int j = 0; j < 4; ++j)
            acc[0][j] = __builtin_amdgcn_wmma_f32_16x16x32_bf16(
                false, a0c.v, false, bc[j].v, (short)0, acc[0][j], false, false);
        for (int j = 0; j < 4; ++j)
            acc[1][j] = __builtin_amdgcn_wmma_f32_16x16x32_bf16(
                false, a1c.v, false, bc[j].v, (short)0, acc[1][j], false, false);
        a0c = a0n; a1c = a1n;
        for (int j = 0; j < 4; ++j) bc[j] = bn[j];
    }
    for (int j = 0; j < 4; ++j)
        acc[0][j] = __builtin_amdgcn_wmma_f32_16x16x32_bf16(
            false, a0c.v, false, bc[j].v, (short)0, acc[0][j], false, false);
    for (int j = 0; j < 4; ++j)
        acc[1][j] = __builtin_amdgcn_wmma_f32_16x16x32_bf16(
            false, a1c.v, false, bc[j].v, (short)0, acc[1][j], false, false);

    for (int mi = 0; mi < 2; ++mi)
        for (int j = 0; j < 4; ++j)
            for (int i = 0; i < 8; ++i) {
                const int row = m0 + mi * 16 + half * 8 + i;   // C rows: i + 8*half
                const int col = n0 + j * 16 + l16;
                C[(size_t)row * N + col] = acc[mi][j][i];
            }
}

// ---------------------------------------------------------------------------
// RoPE: in fp32 [rows, heads*128] -> out bf16, rotate-half formulation.
// ---------------------------------------------------------------------------
__global__ __launch_bounds__(256) void rope_kernel(const float* __restrict__ in,
                                                   unsigned short* __restrict__ out,
                                                   int heads, int total) {
    const int tid = blockIdx.x * 256 + threadIdx.x;
    if (tid >= total) return;
    const int pair = tid & 63;
    const int t2 = tid >> 6;
    const int hh = t2 % heads;
    const int row = t2 / heads;               // b*2048 + s
    const int s = row & 2047;
    const size_t base = (size_t)row * heads * 128 + hh * 128 + pair;
    const float x1 = in[base], x2 = in[base + 64];
    const float inv = __expf(-(float)pair * 0.14391156815f);   // ln(10000)/64
    const float th = (float)s * inv;
    const float cs = __cosf(th), sn = __sinf(th);
    out[base]      = f2bf(x1 * cs - x2 * sn);
    out[base + 64] = f2bf(x2 * cs + x1 * sn);
}

// ---------------------------------------------------------------------------
// V transpose: fp32 [b*2048+s][kv*128+d] -> bf16 vt[b][kv][d][s]
// ---------------------------------------------------------------------------
__global__ __launch_bounds__(256) void vtrans_kernel(const float* __restrict__ v32,
                                                     unsigned short* __restrict__ vt,
                                                     int total) {
    const int tid = blockIdx.x * 256 + threadIdx.x;
    if (tid >= total) return;
    const int srow = tid & 2047;
    const int r = tid >> 11;
    const int d = r & 127;
    const int r2 = r >> 7;                    // b*4 + kvh
    const size_t in = ((size_t)((r2 >> 2) * 2048 + srow)) * 512 + (r2 & 3) * 128 + d;
    vt[tid] = f2bf(v32[in]);
}

// ---------------------------------------------------------------------------
// Flash attention, causal, GQA g=4. One wave per (b, h, 16-row q-tile).
// kv chunks of 32: 8 WMMAs for Q.K^T, 8 WMMAs for P.V, P reshaped via LDS.
// V fragments prefetched before the LDS round-trip so their global latency
// overlaps the softmax VALU work.
// ---------------------------------------------------------------------------
__global__ __launch_bounds__(256) void flash_kernel(const unsigned short* __restrict__ qb,
                                                    const unsigned short* __restrict__ kb,
                                                    const unsigned short* __restrict__ vt,
                                                    unsigned short* __restrict__ attnb) {
    __shared__ __align__(16) unsigned short pbuf[8][16][32];
    const int lane = threadIdx.x & 31, wave = threadIdx.x >> 5;
    const int l16 = lane & 15, half = lane >> 4;
    const int wid = blockIdx.x * 8 + wave;
    const int qt = wid & 127;
    const int h = (wid >> 7) & 15;
    const int b = wid >> 11;
    const int q0 = qt * 16;
    const int kvh = h >> 2;
    const int kb1 = half * 16;

    // Q fragments (A-matrix layout), 4 chunks over D=128
    BF16Frag qf[4];
    const size_t qbase = ((size_t)(b * 2048 + q0 + l16)) * 2048 + h * 128;
    for (int dd = 0; dd < 4; ++dd) {
        const unsigned short* p = qb + qbase + dd * 32 + kb1;
        qf[dd].h[0] = *(const u16x8*)p;
        qf[dd].h[1] = *(const u16x8*)(p + 8);
    }

    float mrow[8], lrow[8], sc[8], mx[8];
    v8f o[8];
    for (int i = 0; i < 8; ++i) { mrow[i] = -1e30f; lrow[i] = 0.0f; }
    for (int j = 0; j < 8; ++j)
        for (int i = 0; i < 8; ++i) o[j][i] = 0.0f;

    const float scale = 0.08838834764831845f;   // 1/sqrt(128)
    const int kend = q0 + 16;

    for (int kc = 0; kc < kend; kc += 32) {
        v8f s0, s1;
        for (int i = 0; i < 8; ++i) { s0[i] = 0.0f; s1[i] = 0.0f; }

        // scores = Q @ K^T over D=128 (4 K-chunks), two 16-col tiles
        for (int dd = 0; dd < 4; ++dd) {
            {
                const int col = kc + l16;
                const unsigned short* p = kb + ((size_t)(b * 2048 + col)) * 512 + kvh * 128 + dd * 32 + half * 16;
                BF16Frag bf; bf.h[0] = *(const u16x8*)p; bf.h[1] = *(const u16x8*)(p + 8);
                s0 = __builtin_amdgcn_wmma_f32_16x16x32_bf16(false, qf[dd].v, false, bf.v, (short)0, s0, false, false);
            }
            {
                const int col = kc + 16 + l16;
                const unsigned short* p = kb + ((size_t)(b * 2048 + col)) * 512 + kvh * 128 + dd * 32 + half * 16;
                BF16Frag bf; bf.h[0] = *(const u16x8*)p; bf.h[1] = *(const u16x8*)(p + 8);
                s1 = __builtin_amdgcn_wmma_f32_16x16x32_bf16(false, qf[dd].v, false, bf.v, (short)0, s1, false, false);
            }
        }

        // prefetch V chunk fragments (independent of softmax / LDS barrier)
        BF16Frag vf[8];
        for (int j = 0; j < 8; ++j) {
            const unsigned short* p = vt + ((size_t)((b * 4 + kvh) * 128 + j * 16 + l16)) * 2048 + kc + half * 16;
            vf[j].h[0] = *(const u16x8*)p;
            vf[j].h[1] = *(const u16x8*)(p + 8);
        }

        // scale, causal mask, row max (16-lane butterfly within each half)
        for (int i = 0; i < 8; ++i) {
            const int row = q0 + half * 8 + i;
            float a0 = s0[i] * scale; if (kc + l16 > row)      a0 = -1e30f;
            float a1 = s1[i] * scale; if (kc + 16 + l16 > row) a1 = -1e30f;
            s0[i] = a0; s1[i] = a1;
            float m2 = fmaxf(a0, a1);
            for (int d = 1; d < 16; d <<= 1) m2 = fmaxf(m2, __shfl_xor(m2, d, 32));
            mx[i] = m2;
        }
        // online softmax update
        for (int i = 0; i < 8; ++i) {
            const float mn = fmaxf(mrow[i], mx[i]);
            sc[i] = __expf(mrow[i] - mn);
            mrow[i] = mn;
            const float p0 = __expf(s0[i] - mn);
            const float p1 = __expf(s1[i] - mn);
            s0[i] = p0; s1[i] = p1;
            float rs = p0 + p1;
            for (int d = 1; d < 16; d <<= 1) rs += __shfl_xor(rs, d, 32);
            lrow[i] = lrow[i] * sc[i] + rs;
        }
        for (int j = 0; j < 8; ++j)
            for (int i = 0; i < 8; ++i) o[j][i] *= sc[i];

        // P: C-layout -> LDS -> A-layout (bf16)
        for (int i = 0; i < 8; ++i) {
            pbuf[wave][half * 8 + i][l16]      = f2bf(s0[i]);
            pbuf[wave][half * 8 + i][16 + l16] = f2bf(s1[i]);
        }
        asm volatile("s_wait_dscnt 0" ::: "memory");
        BF16Frag pa;
        {
            const unsigned short* p = &pbuf[wave][l16][kb1];
            pa.h[0] = *(const u16x8*)p;
            pa.h[1] = *(const u16x8*)(p + 8);
        }

        // O += P @ V_chunk
        for (int j = 0; j < 8; ++j)
            o[j] = __builtin_amdgcn_wmma_f32_16x16x32_bf16(
                false, pa.v, false, vf[j].v, (short)0, o[j], false, false);
    }

    for (int i = 0; i < 8; ++i) lrow[i] = 1.0f / lrow[i];
    for (int j = 0; j < 8; ++j)
        for (int i = 0; i < 8; ++i) {
            const int row = q0 + half * 8 + i;
            attnb[((size_t)(b * 2048 + row)) * 2048 + h * 128 + j * 16 + l16] =
                f2bf(o[j][i] * lrow[i]);
        }
}

// ---------------------------------------------------------------------------
extern "C" void kernel_launch(void* const* d_in, const int* in_sizes, int n_in,
                              void* d_out, int out_size, void* d_ws, size_t ws_size,
                              hipStream_t stream) {
    const float* x  = (const float*)d_in[0];
    const float* wq = (const float*)d_in[1];
    const float* wk = (const float*)d_in[2];
    const float* wv = (const float*)d_in[3];
    const float* wo = (const float*)d_in[4];
    float* out = (float*)d_out;

    const int M = 4096;          // B*S
    char* w = (char*)d_ws;
    unsigned short* xb  = (unsigned short*)w; w += (size_t)M * 2048 * 2;
    unsigned short* wqb = (unsigned short*)w; w += (size_t)2048 * 2048 * 2;
    unsigned short* wkb = (unsigned short*)w; w += (size_t)512 * 2048 * 2;
    unsigned short* wvb = (unsigned short*)w; w += (size_t)512 * 2048 * 2;
    unsigned short* wob = (unsigned short*)w; w += (size_t)2048 * 2048 * 2;
    float* q32 = (float*)w;                   w += (size_t)M * 2048 * 4;
    float* k32 = (float*)w;                   w += (size_t)M * 512 * 4;
    float* v32 = (float*)w;                   w += (size_t)M * 512 * 4;
    unsigned short* qbv   = (unsigned short*)w; w += (size_t)M * 2048 * 2;
    unsigned short* kbv   = (unsigned short*)w; w += (size_t)M * 512 * 2;
    unsigned short* vtv   = (unsigned short*)w; w += (size_t)M * 512 * 2;
    unsigned short* attnb = (unsigned short*)w; w += (size_t)M * 2048 * 2;

    // one-time bf16 conversions (memory-bound, ~60MB total)
    tobf16_kernel<<<(M * 2048 / 4) / 256, 256, 0, stream>>>(x, xb, M * 2048 / 4);
    tobf16_kernel<<<(2048 * 2048 / 4) / 256, 256, 0, stream>>>(wq, wqb, 2048 * 2048 / 4);
    tobf16_kernel<<<(512 * 2048 / 4) / 256, 256, 0, stream>>>(wk, wkb, 512 * 2048 / 4);
    tobf16_kernel<<<(512 * 2048 / 4) / 256, 256, 0, stream>>>(wv, wvb, 512 * 2048 / 4);
    tobf16_kernel<<<(2048 * 2048 / 4) / 256, 256, 0, stream>>>(wo, wob, 2048 * 2048 / 4);

    // QKV projections (bf16 x bf16 -> fp32); block tile = 256M x 64N
    gemm_wmma<<<dim3(2048 / 64, M / 256), 256, 0, stream>>>(xb, wqb, q32, M, 2048, 2048);
    gemm_wmma<<<dim3(512 / 64,  M / 256), 256, 0, stream>>>(xb, wkb, k32, M, 512, 2048);
    gemm_wmma<<<dim3(512 / 64,  M / 256), 256, 0, stream>>>(xb, wvb, v32, M, 512, 2048);

    // RoPE -> bf16
    const int qpairs = M * 16 * 64;
    const int kpairs = M * 4 * 64;
    rope_kernel<<<qpairs / 256, 256, 0, stream>>>(q32, qbv, 16, qpairs);
    rope_kernel<<<kpairs / 256, 256, 0, stream>>>(k32, kbv, 4, kpairs);

    // V -> bf16 transposed [b][kv][d][s]
    const int vtot = M * 512;
    vtrans_kernel<<<vtot / 256, 256, 0, stream>>>(v32, vtv, vtot);

    // causal flash attention (2*16*128 waves, 8 waves/block)
    flash_kernel<<<(2 * 16 * 128) / 8, 256, 0, stream>>>(qbv, kbv, vtv, attnb);

    // output projection -> fp32 d_out
    gemm_wmma<<<dim3(2048 / 64, M / 256), 256, 0, stream>>>(attnb, wob, out, M, 2048, 2048);
}
